// GIN_51007031607812
// MI455X (gfx1250) — matrix operand order repaired and checked by reading, not compiled, not run on hardware
//
#include <hip/hip_runtime.h>
#include <hip/hip_bf16.h>

typedef __attribute__((ext_vector_type(16))) __bf16   v16bf;
typedef __attribute__((ext_vector_type(8)))  float    v8f;
typedef __attribute__((ext_vector_type(8)))  unsigned v8u;
typedef __attribute__((ext_vector_type(4)))  unsigned v4u;

#define BATCH   512
#define N_NODE  200
#define C_IN    200
#define HID     128
#define NTJ     13       // node tiles (13*16 = 208)
#define NTF     13       // C_IN tiles
#define NTH     8        // hidden tiles
#define KC      7        // 6 full K=32 chunks + 1 tail (K 192..207 real, 208..223 zero)
#define STR     216      // row stride in halfs: 432 B (16B aligned, good banking)
#define NTHREADS 512
#define NWAVES   16

// ---- LDS layout (bytes) ----
#define ADJ_BYTES (208 * STR * 2)          // 89856 : adjT bf16 [208][STR]
#define B_BYTES   (208 * STR * 2)          // 89856 : x^T then W1^T bf16
#define C_BYTES   (208 * STR * 2)          // 89856 : M1 bf16
#define H_BYTES   (128 * STR * 2)          // 55296 : H^T bf16
#define OFF_ADJ   0
#define OFF_B     (OFF_ADJ + ADJ_BYTES)    // 89856
#define OFF_C     (OFF_B + B_BYTES)        // 179712
#define OFF_H     (OFF_C + C_BYTES)        // 269568
#define OFF_CS    (OFF_H + H_BYTES)        // 324864 : colsum f32[128]
#define OFF_B1    (OFF_CS + 512)           // 325376 : b1 f32[128]
#define OFF_RED   (OFF_B1 + 512)           // 325888 : red f32[256]
#define LDS_BYTES (OFF_RED + 1024)         // 326912  (<= 327680)

__device__ __forceinline__ unsigned short f2bf(float f) {
    unsigned u = __builtin_bit_cast(unsigned, f);
    unsigned r = (u + 0x7FFFu + ((u >> 16) & 1u)) >> 16;   // RNE
    return (unsigned short)r;
}
__device__ __forceinline__ float bf2f(unsigned short h) {
    unsigned u = ((unsigned)h) << 16;
    return __builtin_bit_cast(float, u);
}

// A-frag 16x32: row = M (lane&15), K contiguous. Two 16B LDS loads.
// VGPR v<4: K = 2v + l16*8 ; v>=4: K = 16 + 2(v-4) + l16*8.
__device__ __forceinline__ v16bf load_fragA(const unsigned short* base, int row,
                                            int kbase, int l16, bool tail) {
    const v4u* p = (const v4u*)(base + row * STR + kbase + l16 * 8);
    v4u lo = p[0];
    v4u hi;
    if (tail) hi = (v4u){0u, 0u, 0u, 0u};       // K 208..223 -> zero (uniform)
    else      hi = p[2];                        // byte offset 32
    v8u r;
    r[0] = lo[0]; r[1] = lo[1]; r[2] = lo[2]; r[3] = lo[3];
    r[4] = hi[0]; r[5] = hi[1]; r[6] = hi[2]; r[7] = hi[3];
    return __builtin_bit_cast(v16bf, r);
}

// B-frag 32x16 stored transposed [n][K], K contiguous. Two 16B LDS loads.
// VGPR v<4: K = 2v + l16*8 ; v>=4: K = 16 + 2(v-4) + l16*8.
__device__ __forceinline__ v16bf load_fragB(const unsigned short* base, int row,
                                            int kbase, int l16, bool tail) {
    const v4u* p = (const v4u*)(base + row * STR + kbase + l16 * 8);
    v4u lo = p[0], hi = p[2];
    v8u r;
    r[0] = lo[0]; r[1] = lo[1]; r[2] = lo[2]; r[3] = lo[3];
    r[4] = hi[0]; r[5] = hi[1]; r[6] = hi[2]; r[7] = hi[3];
    if (tail) {                                 // hi VGPRs would be K 208..223
#pragma unroll
        for (int v = 4; v < 8; ++v) r[v] = 0u;
    }
    return __builtin_bit_cast(v16bf, r);
}

__device__ __forceinline__ v8f wmma_bf16(v16bf a, v16bf b, v8f c) {
    return __builtin_amdgcn_wmma_f32_16x16x32_bf16(false, a, false, b,
                                                   (short)0, c, false, false);
}

__global__ __launch_bounds__(NTHREADS) void gin_fused_kernel(
    const float* __restrict__ x, const int* __restrict__ adj,
    const float* __restrict__ W1, const float* __restrict__ b1,
    const float* __restrict__ W2, const float* __restrict__ b2,
    const float* __restrict__ eps1p, const float* __restrict__ eps2p,
    float* __restrict__ out)
{
    extern __shared__ char smem[];
    unsigned short* AdjH = (unsigned short*)(smem + OFF_ADJ);
    unsigned short* Bh   = (unsigned short*)(smem + OFF_B);
    unsigned short* Ch   = (unsigned short*)(smem + OFF_C);
    unsigned short* Hh   = (unsigned short*)(smem + OFF_H);
    float* colsum = (float*)(smem + OFF_CS);
    float* b1s    = (float*)(smem + OFF_B1);
    float* red    = (float*)(smem + OFF_RED);

    const int b    = blockIdx.x;
    const int tid  = threadIdx.x;
    const int lane = tid & 31;
    const int wave = tid >> 5;
    const int l16  = (lane >> 4) & 1;
    const int n0   = lane & 15;

    const float e1 = 1.0f + *eps1p;
    const float e2 = 1.0f + *eps2p;

    // ---- zero entire LDS (all padding must be exact zeros) ----
    {
        unsigned* z = (unsigned*)smem;
        for (int i = tid; i < LDS_BYTES / 4; i += NTHREADS) z[i] = 0u;
    }
    __syncthreads();

    // ---- stage-in: adjT as bf16 {0,1}, x^T as bf16, b1 ----
    {
        const int*   adjb = adj + (size_t)b * N_NODE * N_NODE;
        const float* xb   = x   + (size_t)b * N_NODE * C_IN;
        for (int idx = tid; idx < N_NODE * N_NODE; idx += NTHREADS) {
            int i = idx / N_NODE, j = idx - i * N_NODE;
            AdjH[j * STR + i] = (unsigned short)((unsigned)adjb[idx] * 0x3F80u);
        }
        for (int idx = tid; idx < N_NODE * C_IN; idx += NTHREADS) {
            int i = idx / C_IN, f = idx - i * C_IN;
            Bh[f * STR + i] = f2bf(xb[idx]);
        }
        if (tid < HID) b1s[tid] = b1[tid];
    }
    __syncthreads();

    // ---- stage 2: agg1 = adjT @ x ; M1 = (1+eps1)*x + agg1 -> Ch, 2x2 blocked ----
    for (int blk = wave; blk < 49; blk += NWAVES) {
        const int jt0 = (blk / 7) * 2, ft0 = (blk % 7) * 2;
        v8f acc[2][2] = {};
#pragma unroll
        for (int kc = 0; kc < KC; ++kc) {
            const bool tail = (kc == KC - 1);
            const int kb = kc * 32;
            v16bf a0 = load_fragA(AdjH, (jt0    ) * 16 + n0, kb, l16, tail);
            v16bf a1 = load_fragA(AdjH, (jt0 + 1) * 16 + n0, kb, l16, tail);
            v16bf b0 = load_fragB(Bh,   (ft0    ) * 16 + n0, kb, l16, tail);
            v16bf b1f= load_fragB(Bh,   (ft0 + 1) * 16 + n0, kb, l16, tail);
            acc[0][0] = wmma_bf16(a0, b0,  acc[0][0]);
            acc[0][1] = wmma_bf16(a0, b1f, acc[0][1]);
            acc[1][0] = wmma_bf16(a1, b0,  acc[1][0]);
            acc[1][1] = wmma_bf16(a1, b1f, acc[1][1]);
        }
#pragma unroll
        for (int p = 0; p < 2; ++p) {
            const int jt = jt0 + p;
            if (jt >= NTJ) continue;
#pragma unroll
            for (int q = 0; q < 2; ++q) {
                const int ft = ft0 + q;
                if (ft >= NTF) continue;
                const int f  = ft * 16 + n0;
                const int j0 = jt * 16 + l16 * 8;
#pragma unroll
                for (int v = 0; v < 8; v += 2) {
                    unsigned xpk = *(const unsigned*)(Bh + f * STR + j0 + v);
                    float m0 = e1 * bf2f((unsigned short)(xpk & 0xFFFFu)) + acc[p][q][v];
                    float m1 = e1 * bf2f((unsigned short)(xpk >> 16))     + acc[p][q][v + 1];
                    Ch[(j0 + v    ) * STR + f] = f2bf(m0);
                    Ch[(j0 + v + 1) * STR + f] = f2bf(m1);
                }
            }
        }
    }
    __syncthreads();

    // ---- stage 3a: overwrite x^T with W1^T (bf16, K-contiguous) ----
    for (int idx = tid; idx < 208 * HID; idx += NTHREADS) {
        const int k = idx >> 7, n = idx & 127;          // coalesced over n
        Bh[n * STR + k] = (k < C_IN) ? f2bf(W1[k * HID + n]) : (unsigned short)0;
    }
    __syncthreads();

    // ---- stage 3b: H = relu(M1 @ W1 + b1) -> H^T, 2x2 blocked ----
    for (int blk = wave; blk < 28; blk += NWAVES) {
        const int jt0 = (blk / 4) * 2, ht0 = (blk % 4) * 2;
        v8f acc[2][2] = {};
#pragma unroll
        for (int kc = 0; kc < KC; ++kc) {
            const bool tail = (kc == KC - 1);
            const int kb = kc * 32;
            v16bf a0 = load_fragA(Ch, (jt0    ) * 16 + n0, kb, l16, tail);
            v16bf a1 = load_fragA(Ch, (jt0 + 1) * 16 + n0, kb, l16, tail);
            v16bf b0 = load_fragB(Bh, (ht0    ) * 16 + n0, kb, l16, tail);
            v16bf b1f= load_fragB(Bh, (ht0 + 1) * 16 + n0, kb, l16, tail);
            acc[0][0] = wmma_bf16(a0, b0,  acc[0][0]);
            acc[0][1] = wmma_bf16(a0, b1f, acc[0][1]);
            acc[1][0] = wmma_bf16(a1, b0,  acc[1][0]);
            acc[1][1] = wmma_bf16(a1, b1f, acc[1][1]);
        }
#pragma unroll
        for (int p = 0; p < 2; ++p) {
            const int jt = jt0 + p;
            if (jt >= NTJ) continue;
#pragma unroll
            for (int q = 0; q < 2; ++q) {
                const int hc = (ht0 + q) * 16 + n0;
                const float bias = b1s[hc];
                const int j0 = jt * 16 + l16 * 8;
#pragma unroll
                for (int v = 0; v < 8; v += 2) {
                    float h0 = acc[p][q][v]     + bias;
                    float h1 = acc[p][q][v + 1] + bias;
                    h0 = (j0 + v     < N_NODE) ? fmaxf(h0, 0.0f) : 0.0f;
                    h1 = (j0 + v + 1 < N_NODE) ? fmaxf(h1, 0.0f) : 0.0f;
                    unsigned pk = (unsigned)f2bf(h0) | ((unsigned)f2bf(h1) << 16);
                    *(unsigned*)(Hh + hc * STR + j0 + v) = pk;
                }
            }
        }
    }
    __syncthreads();

    // ---- stage 4: agg2 = adjT @ H ; colsum[hc] += sum_j ((1+eps2)H + agg2) ----
    for (int blk = wave; blk < 28; blk += NWAVES) {
        const int jt0 = (blk / 4) * 2, ht0 = (blk % 4) * 2;
        v8f acc[2][2] = {};
#pragma unroll
        for (int kc = 0; kc < KC; ++kc) {
            const bool tail = (kc == KC - 1);
            const int kb = kc * 32;
            v16bf a0 = load_fragA(AdjH, (jt0    ) * 16 + n0, kb, l16, tail);
            v16bf a1 = load_fragA(AdjH, (jt0 + 1) * 16 + n0, kb, l16, tail);
            v16bf b0 = load_fragB(Hh,   (ht0    ) * 16 + n0, kb, l16, tail);
            v16bf b1f= load_fragB(Hh,   (ht0 + 1) * 16 + n0, kb, l16, tail);
            acc[0][0] = wmma_bf16(a0, b0,  acc[0][0]);
            acc[0][1] = wmma_bf16(a0, b1f, acc[0][1]);
            acc[1][0] = wmma_bf16(a1, b0,  acc[1][0]);
            acc[1][1] = wmma_bf16(a1, b1f, acc[1][1]);
        }
#pragma unroll
        for (int p = 0; p < 2; ++p) {
            const int jt = jt0 + p;
            if (jt >= NTJ) continue;
#pragma unroll
            for (int q = 0; q < 2; ++q) {
                const int hc = (ht0 + q) * 16 + n0;
                const int j0 = jt * 16 + l16 * 8;
                float s = 0.0f;
#pragma unroll
                for (int v = 0; v < 8; v += 2) {
                    unsigned hpk = *(const unsigned*)(Hh + hc * STR + j0 + v);
                    float h0 = bf2f((unsigned short)(hpk & 0xFFFFu));
                    float h1 = bf2f((unsigned short)(hpk >> 16));
                    s += (j0 + v     < N_NODE) ? (acc[p][q][v]     + e2 * h0) : 0.0f;
                    s += (j0 + v + 1 < N_NODE) ? (acc[p][q][v + 1] + e2 * h1) : 0.0f;
                }
                atomicAdd(&colsum[hc], s);
            }
        }
    }
    __syncthreads();

    // ---- final: out[b,:] = colsum @ W2 / N + b2 ----
    if (tid < HID) {
        const float c = colsum[tid];
        red[tid]       = c * W2[tid * 2 + 0];
        red[128 + tid] = c * W2[tid * 2 + 1];
    }
    __syncthreads();
    if (tid < 2) {
        float s = 0.0f;
        for (int i = 0; i < 128; ++i) s += red[tid * 128 + i];
        out[b * 2 + tid] = s * (1.0f / (float)N_NODE) + b2[tid];
    }
}

extern "C" void kernel_launch(void* const* d_in, const int* in_sizes, int n_in,
                              void* d_out, int out_size, void* d_ws, size_t ws_size,
                              hipStream_t stream) {
    (void)in_sizes; (void)n_in; (void)out_size; (void)d_ws; (void)ws_size;
    const float* x   = (const float*)d_in[0];
    const int*   adj = (const int*)  d_in[1];
    const float* W1  = (const float*)d_in[2];
    const float* b1  = (const float*)d_in[3];
    const float* W2  = (const float*)d_in[4];
    const float* b2  = (const float*)d_in[5];
    const float* e1  = (const float*)d_in[6];
    const float* e2  = (const float*)d_in[7];
    float* out = (float*)d_out;

    static bool attr_set = false;   // idempotent attribute raise for >64KB dyn-LDS
    if (!attr_set) {
        hipFuncSetAttribute((const void*)gin_fused_kernel,
                            hipFuncAttributeMaxDynamicSharedMemorySize, LDS_BYTES);
        attr_set = true;
    }
    hipLaunchKernelGGL(gin_fused_kernel, dim3(BATCH), dim3(NTHREADS), LDS_BYTES,
                       stream, x, adj, W1, b1, W2, b2, e1, e2, out);
}
